// UnionAttention_76433238000453
// MI455X (gfx1250) — matrix operand order repaired and checked by reading, not compiled, not run on hardware
//
#include <hip/hip_runtime.h>

// ---------------------------------------------------------------------------
// UnionAttention for MI455X (gfx1250, wave32).
// Memory-bound: 2 passes over x (32MB @ 23.3TB/s ~ 1.4us) + 67M v_exp_f32.
// BN / channel-mask / percentile-mask folded into per-(b,c) affine so the big
// [B,H,W,C] intermediates are never materialized. The four CxC GEMMs run on
// v_wmma_f32_16x16x32_f16 (f16 in, f32 accumulate), batches packed in M, with
// a zero-padded f16 A matrix in workspace so fragment loads are branch-free.
// ---------------------------------------------------------------------------

#define EPS_ 1e-3f
#define B_   4
#define C_   256
#define HW_  4096          // 64*64
#define NSPLIT 64
#define PPS (HW_/NSPLIT)   // pixels per split = 64

typedef __attribute__((ext_vector_type(16))) _Float16 v16h;
typedef __attribute__((ext_vector_type(8)))  float    v8f;

// workspace layout (float offsets)
#define OFF_PART 0                              // [B][NSPLIT][C] f32
#define OFF_GNH  (OFF_PART + B_*NSPLIT*C_)      // [16][C] f16 (zero-padded A)
#define OFF_Q    (OFF_GNH  + 2048)              // [B][C] f32
#define OFF_K    (OFF_Q    + B_*C_)
#define OFF_V    (OFF_K    + B_*C_)
#define OFF_ATTH (OFF_V    + B_*C_)             // [16][C] f16 (zero-padded A)
#define OFF_CM   (OFF_ATTH + 2048)              // [B][C] f32
#define OFF_WF   (OFF_CM   + B_*C_)             // [B][C][3] folded weights
#define OFF_BIAS (OFF_WF   + B_*C_*3)           // [B][3] (pad 16)
#define OFF_QS   (OFF_BIAS + 16)                // [B][HW]
#define OFF_KS   (OFF_QS   + B_*HW_)
#define OFF_VS   (OFF_KS   + B_*HW_)
#define OFF_KST  (OFF_VS   + B_*HW_)            // [B][2] kmax,kmin

// --- K1: partial sums for global average pool (coalesced over channels) ----
__global__ void k_gap_partial(const float* __restrict__ x, float* __restrict__ part) {
    int b = blockIdx.x / NSPLIT, s = blockIdx.x % NSPLIT, c = threadIdx.x;
    const float* xp = x + ((size_t)(b*HW_ + s*PPS))*C_ + c;
    float acc = 0.f;
    for (int p = 0; p < PPS; ++p) acc += xp[(size_t)p*C_];
    part[(b*NSPLIT + s)*C_ + c] = acc;
}

// --- K2: finish GAP + channel-attention BatchNorm, emit padded f16 A -------
// grid = 16 blocks: rows 0..3 = batches, rows 4..15 written as zeros.
__global__ void k_gap_bn(const float* __restrict__ part,
                         const float* __restrict__ g, const float* __restrict__ be,
                         const float* __restrict__ m, const float* __restrict__ v,
                         _Float16* __restrict__ gnh) {
    int b = blockIdx.x, c = threadIdx.x;
    if (b >= B_) { gnh[b*C_ + c] = (_Float16)0.f; return; }
    float acc = 0.f;
    for (int s = 0; s < NSPLIT; ++s) acc += part[(b*NSPLIT + s)*C_ + c];
    float gap = acc * (1.0f / (float)HW_);
    float gn = (gap - m[c]) * rsqrtf(v[c] + EPS_) * g[c] + be[c];
    gnh[b*C_ + c] = (_Float16)gn;
}

// --- WMMA helper: one 16x16 tile of [16 x 256] @ [256 x 256] ---------------
// A is pre-padded f16 [16][256]. f32 accumulate.
// Layouts used here (wave32; A/B k order is contiguous, not the ISA table order):
//   A frag elem e, lane l: M = l&15,          K = e + 16*(l>>4)
//   B frag elem e, lane l: N = l&15,          K = e + 16*(l>>4)
//   D elem r,      lane l: N = l&15,          M = r + 8*(l>>4)
__device__ __forceinline__ void wmma_tile(const _Float16* __restrict__ A16,
                                          const float* __restrict__ W,
                                          int ntile, float out[8], int lane) {
    v8f acc = {};
    int mrow = lane & 15;
    int lhi  = lane >> 4;
    int ncol = ntile*16 + (lane & 15);
    #pragma unroll
    for (int kt = 0; kt < 8; ++kt) {            // K = 256 = 8 * 32
        v16h a, bf;
        #pragma unroll
        for (int e = 0; e < 16; ++e) {
            int ka = kt*32 + e + 16*lhi;
            a[e] = A16[mrow*C_ + ka];
            int kb = kt*32 + e + 16*lhi;
            bf[e] = (_Float16)W[kb*C_ + ncol];
        }
        acc = __builtin_amdgcn_wmma_f32_16x16x32_f16(
                  false, a, false, bf, (short)0, acc, false, false);
    }
    #pragma unroll
    for (int r = 0; r < 8; ++r) out[r] = acc[r];
}

// --- K3: q,k,v = gn @ {wq,wk,wv} via WMMA (48 tiles, 1 wave each) ----------
__global__ void k_qkv_gemm(const _Float16* __restrict__ gnh,
                           const float* __restrict__ wq, const float* __restrict__ wk,
                           const float* __restrict__ wv,
                           float* __restrict__ q, float* __restrict__ k,
                           float* __restrict__ v) {
    int t = blockIdx.x;                 // 0..47
    int which = t / 16, nt = t % 16;
    int lane = threadIdx.x;
    const float* W = (which == 0) ? wq : (which == 1) ? wk : wv;
    float*       O = (which == 0) ? q  : (which == 1) ? k  : v;
    float out[8];
    wmma_tile(gnh, W, nt, out, lane);
    int lhi = lane >> 4, n = nt*16 + (lane & 15);
    #pragma unroll
    for (int r = 0; r < 8; ++r) { int mm = r + 8*lhi; if (mm < B_) O[mm*C_ + n] = out[r]; }
}

// --- K4: channel attention, head_dim=1 softmax over C; pad f16 A for K5 ----
// grid = 16 blocks: rows >= B_ just zero the padded A rows.
__global__ void k_chan_att(const float* __restrict__ q, const float* __restrict__ k,
                           const float* __restrict__ v, _Float16* __restrict__ atth) {
    __shared__ float sq[C_], sk[C_], sv[C_], smax[C_], smin[C_];
    int b = blockIdx.x, i = threadIdx.x;
    if (b >= B_) { atth[b*C_ + i] = (_Float16)0.f; return; }
    sq[i] = q[b*C_ + i]; sk[i] = k[b*C_ + i]; sv[i] = v[b*C_ + i];
    smax[i] = sk[i]; smin[i] = sk[i];
    __syncthreads();
    for (int off = 128; off > 0; off >>= 1) {
        if (i < off) { smax[i] = fmaxf(smax[i], smax[i+off]); smin[i] = fminf(smin[i], smin[i+off]); }
        __syncthreads();
    }
    float kmax = smax[0], kmin = smin[0];
    float qi = sq[i];
    float mi = (qi >= 0.f) ? qi*kmax : qi*kmin;   // max_j(qi*kj) for stability
    float num = 0.f, den = 0.f;
    for (int j = 0; j < C_; ++j) {
        float e = __expf(qi*sk[j] - mi);
        num += e * sv[j];  den += e;
    }
    atth[b*C_ + i] = (_Float16)(num / den);
}

// --- K5: cmask = sigmoid(att @ wp) via WMMA --------------------------------
__global__ void k_cmask_gemm(const _Float16* __restrict__ atth,
                             const float* __restrict__ wp, float* __restrict__ cm) {
    int nt = blockIdx.x, lane = threadIdx.x;
    float out[8];
    wmma_tile(atth, wp, nt, out, lane);
    int lhi = lane >> 4, n = nt*16 + (lane & 15);
    #pragma unroll
    for (int r = 0; r < 8; ++r) {
        int mm = r + 8*lhi;
        if (mm < B_) cm[mm*C_ + n] = 1.0f / (1.0f + __expf(-out[r]));
    }
}

// --- K6: 10th-percentile (linear: 0.5*(s[25]+s[26])) + fold BN/masks -------
// co*pmask = x*(1+cm)*s_c*pm + t_c*pm  =>  wf[b,c,f], bias[b,f]
__global__ void k_pct_coeff(const float* __restrict__ cm,
                            const float* __restrict__ bng, const float* __restrict__ bnb,
                            const float* __restrict__ bnm, const float* __restrict__ bnv,
                            const float* __restrict__ wqkv,
                            float* __restrict__ wf, float* __restrict__ bias) {
    __shared__ float scm[C_];
    __shared__ float rb0[C_], rb1[C_], rb2[C_];
    __shared__ float s25, s26;
    int b = blockIdx.x, i = threadIdx.x;
    float ci = cm[b*C_ + i];
    scm[i] = ci;
    __syncthreads();
    int r = 0;                                    // stable rank: exact order statistic
    for (int j = 0; j < C_; ++j) {
        float cj = scm[j];
        r += (cj < ci) || (cj == ci && j < i);
    }
    if (r == 25) s25 = ci;
    if (r == 26) s26 = ci;
    __syncthreads();
    float thr = 0.5f * (s25 + s26);               // rank 0.1*(256-1)=25.5, linear interp
    float pm  = (ci > thr) ? ci : 0.f;
    float s   = bng[i] * rsqrtf(bnv[i] + EPS_);
    float t   = bnb[i] - bnm[i] * s;
    float sc  = (1.f + ci) * s * pm;
    float w0 = wqkv[i*3+0], w1 = wqkv[i*3+1], w2 = wqkv[i*3+2];
    wf[(b*C_ + i)*3 + 0] = sc * w0;
    wf[(b*C_ + i)*3 + 1] = sc * w1;
    wf[(b*C_ + i)*3 + 2] = sc * w2;
    rb0[i] = t*pm*w0; rb1[i] = t*pm*w1; rb2[i] = t*pm*w2;
    __syncthreads();
    for (int off = 128; off > 0; off >>= 1) {
        if (i < off) { rb0[i] += rb0[i+off]; rb1[i] += rb1[i+off]; rb2[i] += rb2[i+off]; }
        __syncthreads();
    }
    if (i == 0) { bias[b*3+0] = rb0[0]; bias[b*3+1] = rb1[0]; bias[b*3+2] = rb2[0]; }
}

// --- K7: fused 1x1 conv: qs/ks/vs[b,p] = x[b,p,:] . wf[b,:,f] + bias -------
// One wave per pixel; two b128 loads per lane cover all 256 channels.
__global__ void k_qkv_pixels(const float* __restrict__ x, const float* __restrict__ wf,
                             const float* __restrict__ bias,
                             float* __restrict__ qs, float* __restrict__ ks,
                             float* __restrict__ vs) {
    __shared__ float swf[C_*3];
    __shared__ float sb[3];
    int b = blockIdx.x >> 9;            // 512 blocks per batch
    int pblk = blockIdx.x & 511;
    int tid = threadIdx.x;
    for (int t = tid; t < C_*3; t += 256) swf[t] = wf[b*C_*3 + t];
    if (tid < 3) sb[tid] = bias[b*3 + tid];
    __syncthreads();
    int wave = tid >> 5, lane = tid & 31;
    int p = pblk*8 + wave;
    const float4* xp4 = (const float4*)(x + ((size_t)(b*HW_ + p))*C_);
    float a0 = 0.f, a1 = 0.f, a2 = 0.f;
    #pragma unroll
    for (int kk = 0; kk < 2; ++kk) {
        int c4 = kk*32 + lane;          // float4 index; channels c4*4..c4*4+3
        float4 xv = xp4[c4];
        int c = c4*4;
        a0 += xv.x*swf[(c+0)*3+0] + xv.y*swf[(c+1)*3+0] + xv.z*swf[(c+2)*3+0] + xv.w*swf[(c+3)*3+0];
        a1 += xv.x*swf[(c+0)*3+1] + xv.y*swf[(c+1)*3+1] + xv.z*swf[(c+2)*3+1] + xv.w*swf[(c+3)*3+1];
        a2 += xv.x*swf[(c+0)*3+2] + xv.y*swf[(c+1)*3+2] + xv.z*swf[(c+2)*3+2] + xv.w*swf[(c+3)*3+2];
    }
    #pragma unroll
    for (int off = 16; off > 0; off >>= 1) {
        a0 += __shfl_xor(a0, off, 32);
        a1 += __shfl_xor(a1, off, 32);
        a2 += __shfl_xor(a2, off, 32);
    }
    if (lane == 0) {
        qs[b*HW_ + p] = a0 + sb[0];
        ks[b*HW_ + p] = a1 + sb[1];
        vs[b*HW_ + p] = a2 + sb[2];
    }
}

// --- K8: per-batch kmax/kmin for softmax stabilization ---------------------
__global__ void k_ks_stats(const float* __restrict__ ks, float* __restrict__ kst) {
    __shared__ float smax[256], smin[256];
    int b = blockIdx.x, i = threadIdx.x;
    float mx = -3.0e38f, mn = 3.0e38f;
    for (int j = i; j < HW_; j += 256) {
        float v = ks[b*HW_ + j];
        mx = fmaxf(mx, v); mn = fminf(mn, v);
    }
    smax[i] = mx; smin[i] = mn;
    __syncthreads();
    for (int off = 128; off > 0; off >>= 1) {
        if (i < off) { smax[i] = fmaxf(smax[i], smax[i+off]); smin[i] = fminf(smin[i], smin[i+off]); }
        __syncthreads();
    }
    if (i == 0) { kst[b*2] = smax[0]; kst[b*2+1] = smin[0]; }
}

// --- K9: spatial head_dim=1 softmax attention + residual v -----------------
__global__ void k_spatial_att(const float* __restrict__ qs, const float* __restrict__ ks,
                              const float* __restrict__ vs, const float* __restrict__ kst,
                              float* __restrict__ out) {
    __shared__ float skt[256], svt[256];
    int b = blockIdx.x >> 4;
    int itile = blockIdx.x & 15;
    int tid = threadIdx.x;
    int i = itile*256 + tid;
    float qi = qs[b*HW_ + i];
    float kmax = kst[b*2], kmin = kst[b*2+1];
    float mi = (qi >= 0.f) ? qi*kmax : qi*kmin;
    float num = 0.f, den = 0.f;
    for (int t = 0; t < 16; ++t) {
        __syncthreads();
        skt[tid] = ks[b*HW_ + t*256 + tid];
        svt[tid] = vs[b*HW_ + t*256 + tid];
        __syncthreads();
        #pragma unroll 8
        for (int j = 0; j < 256; ++j) {
            float e = __expf(qi*skt[j] - mi);
            num += e * svt[j];
            den += e;
        }
    }
    out[b*HW_ + i] = vs[b*HW_ + i] + num / den;
}

// ---------------------------------------------------------------------------
extern "C" void kernel_launch(void* const* d_in, const int* in_sizes, int n_in,
                              void* d_out, int out_size, void* d_ws, size_t ws_size,
                              hipStream_t stream) {
    (void)in_sizes; (void)n_in; (void)out_size; (void)ws_size;
    const float* x        = (const float*)d_in[0];
    const float* ca_gamma = (const float*)d_in[1];
    const float* ca_beta  = (const float*)d_in[2];
    const float* ca_mean  = (const float*)d_in[3];
    const float* ca_var   = (const float*)d_in[4];
    const float* wq       = (const float*)d_in[5];
    const float* wk       = (const float*)d_in[6];
    const float* wv       = (const float*)d_in[7];
    const float* wp       = (const float*)d_in[8];
    const float* bn_gamma = (const float*)d_in[9];
    const float* bn_beta  = (const float*)d_in[10];
    const float* bn_mean  = (const float*)d_in[11];
    const float* bn_var   = (const float*)d_in[12];
    const float* w_qkv    = (const float*)d_in[13];
    float* out = (float*)d_out;
    float* w   = (float*)d_ws;
    _Float16* gnh  = (_Float16*)(w + OFF_GNH);
    _Float16* atth = (_Float16*)(w + OFF_ATTH);

    k_gap_partial<<<B_*NSPLIT, C_, 0, stream>>>(x, w + OFF_PART);
    k_gap_bn<<<16, C_, 0, stream>>>(w + OFF_PART, ca_gamma, ca_beta, ca_mean, ca_var,
                                    gnh);
    k_qkv_gemm<<<48, 32, 0, stream>>>(gnh, wq, wk, wv,
                                      w + OFF_Q, w + OFF_K, w + OFF_V);
    k_chan_att<<<16, C_, 0, stream>>>(w + OFF_Q, w + OFF_K, w + OFF_V, atth);
    k_cmask_gemm<<<16, 32, 0, stream>>>(atth, wp, w + OFF_CM);
    k_pct_coeff<<<B_, C_, 0, stream>>>(w + OFF_CM, bn_gamma, bn_beta, bn_mean, bn_var,
                                       w_qkv, w + OFF_WF, w + OFF_BIAS);
    k_qkv_pixels<<<B_*512, 256, 0, stream>>>(x, w + OFF_WF, w + OFF_BIAS,
                                             w + OFF_QS, w + OFF_KS, w + OFF_VS);
    k_ks_stats<<<B_, 256, 0, stream>>>(w + OFF_KS, w + OFF_KST);
    k_spatial_att<<<B_*16, 256, 0, stream>>>(w + OFF_QS, w + OFF_KS, w + OFF_VS,
                                             w + OFF_KST, out);
}
